// AutoformerEncoderLayer_23716809408597
// MI455X (gfx1250) — matrix operand order repaired and checked by reading, not compiled, not run on hardware
//
#include <hip/hip_runtime.h>

// Autoformer encoder layer for MI455X (gfx1250).
// wave32, WMMA bf16 (f32 accum), async global->LDS staging (ASYNCcnt),
// double-buffered GEMM pipeline. Workspace requirement: 112 MB.

#define BATCH  2
#define SEQ    2048
#define DMODEL 1024
#define NHEADS 16
#define DHEAD  64
#define DFF    4096
#define ROWS   (BATCH * SEQ)   // 4096

typedef __attribute__((ext_vector_type(8)))  float  v8f;
typedef __attribute__((ext_vector_type(16))) __bf16 v16bf;
typedef __attribute__((ext_vector_type(8)))  __bf16 v8bf;
typedef int v4i __attribute__((vector_size(16)));   // builtin's 16B chunk type

__device__ __forceinline__ v8f wmma_bf16(v16bf a, v16bf b, v8f c) {
  // D = A(16x32 bf16) * B(32x16 bf16) + C(16x16 f32)
  return __builtin_amdgcn_wmma_f32_16x16x32_bf16(false, a, false, b,
                                                 (short)0, c, false, false);
}

// ---- gfx1250 async global->LDS copy (16B per lane), ASYNCcnt tracked ------
__device__ __forceinline__ void async_copy16(const __bf16* g, __bf16* l) {
#if __has_builtin(__builtin_amdgcn_global_load_async_to_lds_b128)
  __builtin_amdgcn_global_load_async_to_lds_b128(
      (__attribute__((address_space(1))) v4i*)(g),
      (__attribute__((address_space(3))) v4i*)(l), 0, 0);
#else
  unsigned lo = (unsigned)(unsigned long long)
      ((__attribute__((address_space(3))) __bf16*)l);
  asm volatile("global_load_async_to_lds_b128 %0, %1, off"
               :: "v"(lo), "v"(g) : "memory");
#endif
}

__device__ __forceinline__ void wait_async0() {
#if __has_builtin(__builtin_amdgcn_s_wait_asynccnt)
  __builtin_amdgcn_s_wait_asynccnt(0);
#else
  asm volatile("s_wait_asynccnt 0x0" ::: "memory");
#endif
}

// ---------------------------------------------------------------------------
// f32 -> bf16 elementwise convert (n multiple of 4)
// ---------------------------------------------------------------------------
__global__ void cvt_kernel(const float* __restrict__ in,
                           __bf16* __restrict__ out, int n)
{
  const int i = (blockIdx.x * blockDim.x + threadIdx.x) * 4;
  if (i >= n) return;
  const float4 v = *(const float4*)(in + i);
  out[i + 0] = (__bf16)v.x;  out[i + 1] = (__bf16)v.y;
  out[i + 2] = (__bf16)v.z;  out[i + 3] = (__bf16)v.w;
}

// ---------------------------------------------------------------------------
// GEMM: C[M,N] = A[M,K] @ W[N,K]^T + bias[N], bf16 inputs, f32 accumulate.
// MODE 0: bias; MODE 1: bias + f32 residual; MODE 2: bias + exact GELU.
// OUTBF: 1 -> write bf16, 0 -> write f32.
// 128x128 macro-tile, 8 waves (2x4), double-buffered async LDS staging.
// ---------------------------------------------------------------------------
template<int MODE, int OUTBF>
__global__ __launch_bounds__(256) void gemm_bias_kernel(
    const __bf16* __restrict__ A, const __bf16* __restrict__ W,
    const float* __restrict__ bias, const float* __restrict__ resid,
    void* __restrict__ Cout, int M, int N, int K)
{
  __shared__ __align__(32) __bf16 Als[2][128][32];
  __shared__ __align__(32) __bf16 Bls[2][128][32];

  const int tid  = threadIdx.x;
  const int lane = tid & 31;
  const int wave = tid >> 5;
  const int half = lane >> 4;
  const int l16  = lane & 15;
  const int wm   = wave >> 2;   // 0..1 -> 64 rows
  const int wn   = wave & 3;    // 0..3 -> 32 cols
  const int m0   = blockIdx.y * 128;
  const int n0   = blockIdx.x * 128;

  v8f acc[4][2];
  #pragma unroll
  for (int i = 0; i < 4; ++i)
    #pragma unroll
    for (int j = 0; j < 2; ++j)
      #pragma unroll
      for (int e = 0; e < 8; ++e) acc[i][j][e] = 0.0f;

  // stage one 128x32 bf16 tile pair into buffer `bf` via async copies:
  // 512 16B chunks per tile, 2 chunks per thread per tile.
#define STAGE_TILE(bf, kkv)                                                  \
  do {                                                                       \
    const __bf16* Ab_ = A + (size_t)m0 * K + (kkv);                          \
    const __bf16* Wb_ = W + (size_t)n0 * K + (kkv);                          \
    _Pragma("unroll")                                                        \
    for (int r_ = 0; r_ < 2; ++r_) {                                         \
      const int c_   = tid + 256 * r_;   /* 0..511 */                        \
      const int row_ = c_ >> 2;                                              \
      const int co_  = (c_ & 3) * 8;     /* bf16 elements, 16B chunks */     \
      async_copy16(Ab_ + (size_t)row_ * K + co_, &Als[bf][row_][co_]);       \
      async_copy16(Wb_ + (size_t)row_ * K + co_, &Bls[bf][row_][co_]);       \
    }                                                                        \
  } while (0)

  STAGE_TILE(0, 0);
  const int nIter = K >> 5;
  for (int i = 0; i < nIter; ++i) {
    const int cur = i & 1;
    wait_async0();      // tile i resident (per-wave ASYNCcnt, loads in order)
    __syncthreads();    // all waves' async portions visible; buf cur^1 free
    if (i + 1 < nIter) STAGE_TILE(cur ^ 1, (i + 1) << 5);  // overlap w/ WMMA

    // A fragment (16-bit A 16x32): element i -> K = i + 16*half
    v16bf afr[4], bfr[2];
    #pragma unroll
    for (int mt = 0; mt < 4; ++mt) {
      const int row = wm * 64 + mt * 16 + l16;
      v8bf lo = *(const v8bf*)&Als[cur][row][half * 16];
      v8bf hi = *(const v8bf*)&Als[cur][row][half * 16 + 8];
      afr[mt] = __builtin_shufflevector(lo, hi, 0,1,2,3,4,5,6,7,8,9,10,11,12,13,14,15);
    }
    // B fragment (32x16): lane = col, K = i + 16*half (contiguous 32B)
    #pragma unroll
    for (int nt = 0; nt < 2; ++nt) {
      const int n = wn * 32 + nt * 16 + l16;
      bfr[nt] = *(const v16bf*)&Bls[cur][n][half * 16];
    }
    #pragma unroll
    for (int mt = 0; mt < 4; ++mt)
      #pragma unroll
      for (int nt = 0; nt < 2; ++nt)
        acc[mt][nt] = wmma_bf16(afr[mt], bfr[nt], acc[mt][nt]);
  }
#undef STAGE_TILE

  // Epilogue. C layout: element j -> row M = j + 8*half, col N = l16.
  #pragma unroll
  for (int mt = 0; mt < 4; ++mt) {
    #pragma unroll
    for (int nt = 0; nt < 2; ++nt) {
      const int col = n0 + wn * 32 + nt * 16 + l16;
      const float bv = bias[col];
      #pragma unroll
      for (int j = 0; j < 8; ++j) {
        const int row = m0 + wm * 64 + mt * 16 + half * 8 + j;
        float vv = acc[mt][nt][j] + bv;
        if (MODE == 1) vv += resid[(size_t)row * N + col];
        if (MODE == 2) vv = 0.5f * vv * (1.0f + erff(vv * 0.70710678118654752f));
        if (OUTBF) ((__bf16*)Cout)[(size_t)row * N + col] = (__bf16)vv;
        else       ((float*)Cout)[(size_t)row * N + col]  = vv;
      }
    }
  }
}

// ---------------------------------------------------------------------------
// Flash attention: softmax(QK^T/8 - 0.1|i-j|) @ V per (batch, head).
// 128 threads = 4 waves, each wave owns 16 query rows; key blocks of 32.
// q/k/v/ao are bf16; softmax state fp32.
// ---------------------------------------------------------------------------
__global__ __launch_bounds__(128) void attn_kernel(
    const __bf16* __restrict__ q, const __bf16* __restrict__ k,
    const __bf16* __restrict__ v, __bf16* __restrict__ ao)
{
  __shared__ __align__(32) __bf16 Kls[32][64];      // [key][dk] (async filled)
  __shared__ __align__(32) __bf16 VlsT[64][32];     // [dk][key] (transposed)
  __shared__ __align__(32) __bf16 Pls[4][16][32];   // per-wave P staging

  const int tid  = threadIdx.x;
  const int lane = tid & 31;
  const int wave = tid >> 5;
  const int half = lane >> 4;
  const int l16  = lane & 15;
  const int b    = blockIdx.x >> 4;
  const int h    = blockIdx.x & 15;
  const int q0   = blockIdx.y * 64 + wave * 16;
  const size_t rbase = (size_t)b * SEQ;

  // Q fragments: 16 rows x dk=64 -> two A fragments (K 0..31, 32..63)
  v16bf qa0, qa1;
  {
    const __bf16* qp = q + (rbase + q0 + l16) * DMODEL + h * DHEAD;
    #pragma unroll
    for (int i = 0; i < 16; ++i) {
      const int kkA = i + half * 16;
      qa0[i] = qp[kkA];
      qa1[i] = qp[kkA + 32];
    }
  }

  float m_[8], l_[8];
  v8f O[4];
  #pragma unroll
  for (int j = 0; j < 8; ++j) { m_[j] = -1e30f; l_[j] = 0.0f; }
  #pragma unroll
  for (int c = 0; c < 4; ++c)
    #pragma unroll
    for (int e = 0; e < 8; ++e) O[c][e] = 0.0f;

  for (int kb = 0; kb < SEQ; kb += 32) {
    __syncthreads();   // previous iteration's LDS readers done
    // K tile via async-to-LDS: 32 rows x 8 chunks(16B) = 256, 2 per thread
    #pragma unroll
    for (int r = 0; r < 2; ++r) {
      const int c   = tid + 128 * r;
      const int row = c >> 3;
      const int co  = (c & 7) * 8;
      async_copy16(k + (rbase + kb + row) * DMODEL + h * DHEAD + co,
                   &Kls[row][co]);
    }
    // V^T scalar restage (transpose): 2048 elems, 16 per thread
    for (int t = tid; t < 32 * 64; t += 128) {
      const int key = t >> 6, d = t & 63;
      VlsT[d][key] = v[(rbase + kb + key) * DMODEL + h * DHEAD + d];
    }
    wait_async0();
    __syncthreads();

    // S = Q @ K_blk^T, two 16-key halves, chained over dk (2 x K=32)
    v8f sc[2];
    #pragma unroll
    for (int nh = 0; nh < 2; ++nh) {
      const int kcol = nh * 16 + l16;
      v16bf b0 = *(const v16bf*)&Kls[kcol][half * 16];
      v16bf b1 = *(const v16bf*)&Kls[kcol][32 + half * 16];
      v8f s;
      #pragma unroll
      for (int e = 0; e < 8; ++e) s[e] = 0.0f;
      s = wmma_bf16(qa0, b0, s);
      s = wmma_bf16(qa1, b1, s);
      sc[nh] = s;
    }

    // Online softmax per query row (row lives across a 16-lane half)
    #pragma unroll
    for (int j = 0; j < 8; ++j) {
      const int qi = q0 + half * 8 + j;
      float s0 = sc[0][j] * 0.125f - 0.1f * fabsf((float)(qi - (kb + l16)));
      float s1 = sc[1][j] * 0.125f - 0.1f * fabsf((float)(qi - (kb + 16 + l16)));
      float mx = fmaxf(s0, s1);
      #pragma unroll
      for (int off = 8; off >= 1; off >>= 1)
        mx = fmaxf(mx, __shfl_xor(mx, off, 32));
      const float mn = fmaxf(m_[j], mx);
      const float p0 = __expf(s0 - mn);
      const float p1 = __expf(s1 - mn);
      float ps = p0 + p1;
      #pragma unroll
      for (int off = 8; off >= 1; off >>= 1)
        ps += __shfl_xor(ps, off, 32);
      const float co = __expf(m_[j] - mn);
      l_[j] = l_[j] * co + ps;
      m_[j] = mn;
      #pragma unroll
      for (int c = 0; c < 4; ++c) O[c][j] *= co;
      Pls[wave][half * 8 + j][l16]      = (__bf16)p0;
      Pls[wave][half * 8 + j][16 + l16] = (__bf16)p1;
    }
    __syncthreads();

    // O += P(16x32) @ V(32x64)
    v8bf plo = *(const v8bf*)&Pls[wave][l16][half * 16];
    v8bf phi = *(const v8bf*)&Pls[wave][l16][half * 16 + 8];
    v16bf pa = __builtin_shufflevector(plo, phi, 0,1,2,3,4,5,6,7,8,9,10,11,12,13,14,15);
    #pragma unroll
    for (int c = 0; c < 4; ++c) {
      v16bf bvf = *(const v16bf*)&VlsT[c * 16 + l16][half * 16];
      O[c] = wmma_bf16(pa, bvf, O[c]);
    }
  }

  #pragma unroll
  for (int c = 0; c < 4; ++c)
    #pragma unroll
    for (int j = 0; j < 8; ++j) {
      const int qi = q0 + half * 8 + j;
      ao[(rbase + qi) * DMODEL + h * DHEAD + c * 16 + l16] =
          (__bf16)(O[c][j] / l_[j]);
    }
}

// ---------------------------------------------------------------------------
// AvgPool1d over L: kernel=25, stride=1, pad=12, count_include_pad=True.
// ---------------------------------------------------------------------------
__global__ void movavg_kernel(const float* __restrict__ x, float* __restrict__ out)
{
  const int idx = blockIdx.x * blockDim.x + threadIdx.x;
  const int total = ROWS * DMODEL;
  if (idx >= total) return;
  const int d = idx % DMODEL;
  const int t = (idx / DMODEL) % SEQ;
  const int b = idx / (DMODEL * SEQ);
  int lo = t - 12; if (lo < 0) lo = 0;
  int hi = t + 12; if (hi > SEQ - 1) hi = SEQ - 1;
  float s = 0.0f;
  for (int u = lo; u <= hi; ++u)
    s += x[((size_t)b * SEQ + u) * DMODEL + d];
  out[idx] = s * (1.0f / 25.0f);
}

// ---------------------------------------------------------------------------
// LayerNorm over D=1024; optional secondary bf16 output for WMMA consumers.
// ---------------------------------------------------------------------------
template<bool DUAL>
__global__ __launch_bounds__(256) void layernorm_kernel(
    const float* __restrict__ x, const float* __restrict__ g,
    const float* __restrict__ be, float* __restrict__ out,
    __bf16* __restrict__ outbf)
{
  __shared__ float redS[8], redQ[8];
  const size_t row = blockIdx.x;
  const float* xr = x + row * DMODEL;
  const int tid = threadIdx.x;
  float v4[4];
  float s = 0.0f, s2 = 0.0f;
  #pragma unroll
  for (int i = 0; i < 4; ++i) {
    const float vv = xr[tid + i * 256];
    v4[i] = vv; s += vv; s2 += vv * vv;
  }
  #pragma unroll
  for (int off = 16; off >= 1; off >>= 1) {
    s  += __shfl_xor(s,  off, 32);
    s2 += __shfl_xor(s2, off, 32);
  }
  if ((tid & 31) == 0) { redS[tid >> 5] = s; redQ[tid >> 5] = s2; }
  __syncthreads();
  float ts = 0.0f, tq = 0.0f;
  #pragma unroll
  for (int w = 0; w < 8; ++w) { ts += redS[w]; tq += redQ[w]; }
  const float mu  = ts * (1.0f / DMODEL);
  const float var = tq * (1.0f / DMODEL) - mu * mu;
  const float rs  = rsqrtf(var + 1e-5f);
  #pragma unroll
  for (int i = 0; i < 4; ++i) {
    const int d = tid + i * 256;
    const float o = (v4[i] - mu) * rs * g[d] + be[d];
    out[row * DMODEL + d] = o;
    if (DUAL) outbf[row * DMODEL + d] = (__bf16)o;
  }
}

// ---------------------------------------------------------------------------
extern "C" void kernel_launch(void* const* d_in, const int* in_sizes, int n_in,
                              void* d_out, int out_size, void* d_ws, size_t ws_size,
                              hipStream_t stream)
{
  const float* x   = (const float*)d_in[0];
  const float* Wq  = (const float*)d_in[1];
  const float* bq  = (const float*)d_in[2];
  const float* Wk  = (const float*)d_in[3];
  const float* bk  = (const float*)d_in[4];
  const float* Wv  = (const float*)d_in[5];
  const float* bv  = (const float*)d_in[6];
  const float* Wo  = (const float*)d_in[7];
  const float* bo  = (const float*)d_in[8];
  const float* W1  = (const float*)d_in[9];
  const float* b1  = (const float*)d_in[10];
  const float* W2  = (const float*)d_in[11];
  const float* b2  = (const float*)d_in[12];
  const float* g1  = (const float*)d_in[13];
  const float* be1 = (const float*)d_in[14];
  const float* g2  = (const float*)d_in[15];
  const float* be2 = (const float*)d_in[16];

  char* ws = (char*)d_ws;
  const size_t MB = 1024 * 1024;
  // bf16 region
  __bf16* xbf  = (__bf16*)(ws + 0 * MB);   // 8 MB; reused later as xn1bf
  __bf16* Wqb  = (__bf16*)(ws + 8 * MB);   // 2 MB
  __bf16* Wkb  = (__bf16*)(ws + 10 * MB);  // 2 MB
  __bf16* Wvb  = (__bf16*)(ws + 12 * MB);  // 2 MB
  __bf16* Wob  = (__bf16*)(ws + 14 * MB);  // 2 MB
  __bf16* W1b  = (__bf16*)(ws + 16 * MB);  // 8 MB
  __bf16* W2b  = (__bf16*)(ws + 24 * MB);  // 8 MB
  __bf16* qb   = (__bf16*)(ws + 32 * MB);  // 8 MB
  __bf16* kb   = (__bf16*)(ws + 40 * MB);  // 8 MB
  __bf16* vb   = (__bf16*)(ws + 48 * MB);  // 8 MB
  __bf16* aob  = (__bf16*)(ws + 56 * MB);  // 8 MB
  __bf16* hb   = (__bf16*)(ws + 32 * MB);  // 32 MB, overlays q/k/v/ao (dead)
  __bf16* xn1b = xbf;                      // overlays xbf (dead after QKV)
  // f32 region
  float* y1  = (float*)(ws + 64 * MB);     // 16 MB; reused as y2
  float* t1  = (float*)(ws + 80 * MB);     // 16 MB; reused as t2
  float* xn1 = (float*)(ws + 96 * MB);     // 16 MB   (total 112 MB)
  float* y2 = y1;
  float* t2 = t1;

  const dim3 blk(256);
  const dim3 gD(DMODEL / 128, ROWS / 128);   // (8, 32)
  const dim3 gF(DFF / 128,    ROWS / 128);   // (32, 32)
  const int SD = ROWS * DMODEL;              // 4M
  const int SW = DMODEL * DMODEL;            // 1M
  const int SF = DFF * DMODEL;               // 4M

  // one-time bf16 conversions (amortized across all WMMA reuse)
  cvt_kernel<<<dim3(SD / 1024), blk, 0, stream>>>(x,  xbf, SD);
  cvt_kernel<<<dim3(SW / 1024), blk, 0, stream>>>(Wq, Wqb, SW);
  cvt_kernel<<<dim3(SW / 1024), blk, 0, stream>>>(Wk, Wkb, SW);
  cvt_kernel<<<dim3(SW / 1024), blk, 0, stream>>>(Wv, Wvb, SW);
  cvt_kernel<<<dim3(SW / 1024), blk, 0, stream>>>(Wo, Wob, SW);
  cvt_kernel<<<dim3(SF / 1024), blk, 0, stream>>>(W1, W1b, SF);
  cvt_kernel<<<dim3(SF / 1024), blk, 0, stream>>>(W2, W2b, SF);

  // QKV projections (bf16 out, consumed by attention WMMA)
  gemm_bias_kernel<0,1><<<gD, blk, 0, stream>>>(xbf, Wqb, bq, nullptr, qb, ROWS, DMODEL, DMODEL);
  gemm_bias_kernel<0,1><<<gD, blk, 0, stream>>>(xbf, Wkb, bk, nullptr, kb, ROWS, DMODEL, DMODEL);
  gemm_bias_kernel<0,1><<<gD, blk, 0, stream>>>(xbf, Wvb, bv, nullptr, vb, ROWS, DMODEL, DMODEL);
  // flash attention (never materializes L x L scores)
  attn_kernel<<<dim3(BATCH * NHEADS, SEQ / 64), dim3(128), 0, stream>>>(qb, kb, vb, aob);
  // output projection + residual: y1 = x + ao @ Wo^T + bo   (f32 out)
  gemm_bias_kernel<1,0><<<gD, blk, 0, stream>>>(aob, Wob, bo, x, y1, ROWS, DMODEL, DMODEL);
  // series decomposition (trend) + layernorm 1 (f32 + bf16 mirror)
  movavg_kernel<<<dim3((ROWS * DMODEL) / 256), blk, 0, stream>>>(y1, t1);
  layernorm_kernel<true><<<dim3(ROWS), blk, 0, stream>>>(t1, g1, be1, xn1, xn1b);
  // FFN: h = gelu(xn1 @ W1^T + b1) (bf16); y2 = xn1 + h @ W2^T + b2 (f32)
  gemm_bias_kernel<2,1><<<gF, blk, 0, stream>>>(xn1b, W1b, b1, nullptr, hb, ROWS, DFF, DMODEL);
  gemm_bias_kernel<1,0><<<gD, blk, 0, stream>>>(hb, W2b, b2, xn1, y2, ROWS, DMODEL, DFF);
  // decomposition + layernorm 2 -> output
  movavg_kernel<<<dim3((ROWS * DMODEL) / 256), blk, 0, stream>>>(y2, t2);
  layernorm_kernel<false><<<dim3(ROWS), blk, 0, stream>>>(t2, g2, be2, (float*)d_out, nullptr);
}
